// AnyUp_16157666968361
// MI455X (gfx1250) — matrix operand order repaired and checked by reading, not compiled, not run on hardware
//
#include <hip/hip_runtime.h>
#include <hip/hip_bf16.h>
#include <stdint.h>

typedef __attribute__((ext_vector_type(16))) __bf16 v16bf;
typedef __attribute__((ext_vector_type(8)))  float  v8f;

#define EPSF 1e-5f
#define PI_F 3.14159265358979f

union FragB { uint32_t u[8]; v16bf v; };

__device__ __forceinline__ uint32_t pack2bf(float a, float b) {
  uint32_t ua = __float_as_uint(a);
  uint32_t ub = __float_as_uint(b);
  ua += 0x7fffu + ((ua >> 16) & 1u);   // round-to-nearest-even bf16
  ub += 0x7fffu + ((ub >> 16) & 1u);
  return (ua >> 16) | (ub & 0xffff0000u);
}

__device__ __forceinline__ int refl(int i, int n) {
  if (i < 0) i = -i;
  if (i >= n) i = 2 * n - 2 - i;
  return i;
}

// A-fragment K index for element-pair e, lane-half h (contiguous k order, k = 16h + 2e; B uses the same)
__device__ __forceinline__ int a_kbase(int e, int h) {
  return 2 * e + 16 * h;
}

// ---------------------------------------------------------------------------
// conv1x1 (3->128) on NCHW image -> [B, P, 128]
// ---------------------------------------------------------------------------
__global__ __launch_bounds__(256) void k_preimg(const float* __restrict__ img,
                                                const float* __restrict__ Wp,
                                                float* __restrict__ Y) {
  int gid = blockIdx.x * 256 + threadIdx.x;     // 2*65536*128
  int o = gid & 127;
  int p = (gid >> 7) & 65535;
  int b = gid >> 23;
  float a = img[(size_t)(b * 3 + 0) * 65536 + p] * Wp[o * 3 + 0]
          + img[(size_t)(b * 3 + 1) * 65536 + p] * Wp[o * 3 + 1]
          + img[(size_t)(b * 3 + 2) * 65536 + p] * Wp[o * 3 + 2];
  Y[(size_t)(b * 65536 + p) * 128 + o] = a;
}

// ---------------------------------------------------------------------------
// Generic 128x128 conv1x1 GEMM with bf16 WMMA, fp32 accum.
// act=1: apply GroupNorm(8)+SiLU to the input while staging A.
// Optional residual add; output row stride/offset for strided concat writes.
// ---------------------------------------------------------------------------
__global__ __launch_bounds__(256) void k_gemm(
    const float* __restrict__ X, const float* __restrict__ W,
    const float* __restrict__ R, float* __restrict__ Y,
    const float* __restrict__ gmean, const float* __restrict__ gvar,
    const float* __restrict__ gamma, const float* __restrict__ beta,
    int P, int act, int outStride, int outOffset) {
  __shared__ uint32_t lA[8192];   // 32KB: A frags [w][s][lane][e]
  __shared__ uint32_t lB[8192];   // 32KB: B frags [t][s][lane][e]
  const int tid = threadIdx.x;
  const int r0 = blockIdx.x * 128;          // global row base (rows = b*P + p)
  const int b  = r0 / P;                    // whole block in one batch (128|P)
  const int bg = b * 8;

  // Stage weights W[n][k] into B-fragment layout (bf16 packed pairs).
  for (int i = tid; i < 8192; i += 256) {
    int e = i & 7, lane = (i >> 3) & 31, s = (i >> 8) & 3, t = i >> 10;
    int h = lane >> 4;
    int n = t * 16 + (lane & 15);
    int k = s * 32 + 2 * e + 16 * h;
    lB[i] = pack2bf(W[n * 128 + k], W[n * 128 + k + 1]);
  }
  // Stage activations into A-fragment layout (optionally GN+SiLU fused).
  for (int i = tid; i < 8192; i += 256) {
    int e = i & 7, lane = (i >> 3) & 31, s = (i >> 8) & 3, w = i >> 10;
    int h = lane >> 4;
    int row = r0 + w * 16 + (lane & 15);
    int kb = s * 32 + a_kbase(e, h);
    float x0 = X[(size_t)row * 128 + kb];
    float x1 = X[(size_t)row * 128 + kb + 1];
    if (act) {
      int g = kb >> 4;                       // kb even => same group for kb+1
      float m = gmean[bg + g];
      float rstd = rsqrtf(gvar[bg + g] + EPSF);
      float n0 = (x0 - m) * rstd * gamma[kb] + beta[kb];
      float n1 = (x1 - m) * rstd * gamma[kb + 1] + beta[kb + 1];
      x0 = n0 / (1.f + __expf(-n0));
      x1 = n1 / (1.f + __expf(-n1));
    }
    lA[i] = pack2bf(x0, x1);
  }
  __syncthreads();

  const int wave = tid >> 5, lane = tid & 31;
  v8f acc[8];
  #pragma unroll
  for (int t = 0; t < 8; ++t)
    #pragma unroll
    for (int v = 0; v < 8; ++v) acc[t][v] = 0.f;

  #pragma unroll
  for (int s = 0; s < 4; ++s) {
    FragB fa;
    #pragma unroll
    for (int e = 0; e < 8; ++e) fa.u[e] = lA[(wave * 4 + s) * 256 + lane * 8 + e];
    #pragma unroll
    for (int t = 0; t < 8; ++t) {
      FragB fb;
      #pragma unroll
      for (int e = 0; e < 8; ++e) fb.u[e] = lB[(t * 4 + s) * 256 + lane * 8 + e];
      acc[t] = __builtin_amdgcn_wmma_f32_16x16x32_bf16(
          false, fa.v, false, fb.v, (short)0, acc[t], false, false);
    }
  }

  const int h = lane >> 4;
  #pragma unroll
  for (int t = 0; t < 8; ++t) {
    int n = t * 16 + (lane & 15);
    #pragma unroll
    for (int v = 0; v < 8; ++v) {
      int row = r0 + wave * 16 + v + 8 * h;
      float val = acc[t][v];
      if (R) val += R[(size_t)row * 128 + n];
      Y[(size_t)row * outStride + outOffset + n] = val;
    }
  }
}

// ---------------------------------------------------------------------------
// GroupNorm statistics: one block per (batch, group)
// ---------------------------------------------------------------------------
__global__ __launch_bounds__(256) void k_gnstats(const float* __restrict__ H,
                                                 float* __restrict__ mean,
                                                 float* __restrict__ var, int P) {
  __shared__ float rs[256], rq[256];
  int tid = threadIdx.x;
  int b = blockIdx.x >> 3, g = blockIdx.x & 7;
  float s = 0.f, q = 0.f;
  int total = P * 16;
  for (int i = tid; i < total; i += 256) {
    int p = i >> 4, c = (g << 4) + (i & 15);
    float x = H[((size_t)b * P + p) * 128 + c];
    s += x; q += x * x;
  }
  rs[tid] = s; rq[tid] = q;
  __syncthreads();
  for (int o = 128; o > 0; o >>= 1) {
    if (tid < o) { rs[tid] += rs[tid + o]; rq[tid] += rq[tid + o]; }
    __syncthreads();
  }
  if (tid == 0) {
    float n = (float)total;
    float m = rs[0] / n;
    mean[blockIdx.x] = m;
    var[blockIdx.x] = rq[0] / n - m * m;
  }
}

// ---------------------------------------------------------------------------
// Axial 2D RoPE, in place on [B, P, 128]
// ---------------------------------------------------------------------------
__global__ __launch_bounds__(256) void k_rope(float* __restrict__ E) {
  int gid = blockIdx.x * 256 + threadIdx.x;  // 2*65536*64
  int i = gid & 63;
  int p = (gid >> 6) & 65535;
  int b = gid >> 22;
  int y = p >> 8, x = p & 255;
  int j = i & 31;
  float freq = __powf(100.f, (float)j * (1.f / 32.f));
  float pos = (i < 32) ? ((float)y + 0.5f) * (1.f / 256.f)
                       : ((float)x + 0.5f) * (1.f / 256.f);
  float ang = pos * freq * PI_F;
  float s, c;
  __sincosf(ang, &s, &c);
  size_t base = ((size_t)(b * 65536 + p)) * 128 + ((i < 32) ? 0 : 64) + 2 * j;
  float v0 = E[base], v1 = E[base + 1];
  E[base] = v0 * c - v1 * s;
  E[base + 1] = v0 * s + v1 * c;
}

// ---------------------------------------------------------------------------
// 8x8 adaptive avg pool [B,P,128] -> kcat[:, :, 0:128] (row stride 256)
// ---------------------------------------------------------------------------
__global__ __launch_bounds__(256) void k_pool(const float* __restrict__ X,
                                              float* __restrict__ kcat) {
  int gid = blockIdx.x * 256 + threadIdx.x;  // 2*1024*128
  int c = gid & 127;
  int cell = (gid >> 7) & 1023;
  int b = gid >> 17;
  int cy = cell >> 5, cx = cell & 31;
  float s = 0.f;
  for (int dy = 0; dy < 8; ++dy)
    for (int dx = 0; dx < 8; ++dx) {
      int p = (cy * 8 + dy) * 256 + cx * 8 + dx;
      s += X[(size_t)(b * 65536 + p) * 128 + c];
    }
  kcat[(size_t)(b * 1024 + cell) * 256 + c] = s * (1.f / 64.f);
}

// ---------------------------------------------------------------------------
// channel-normalized feature mean: fmean[b,p] = mean_c(f/max(||f||,1e-12))
// ---------------------------------------------------------------------------
__global__ __launch_bounds__(256) void k_featmean(const float* __restrict__ F,
                                                  float* __restrict__ fmean) {
  int gid = blockIdx.x * 256 + threadIdx.x;
  if (gid >= 2048) return;
  int b = gid >> 10, p = gid & 1023;
  float ss = 0.f, sm = 0.f;
  for (int c = 0; c < 384; ++c) {
    float v = F[(size_t)(b * 384 + c) * 1024 + p];
    ss += v * v; sm += v;
  }
  float nrm = fmaxf(sqrtf(ss), 1e-12f);
  fmean[gid] = sm / nrm * (1.f / 384.f);
}

// ---------------------------------------------------------------------------
// LFU: 5x5 reflect conv 1->128 channels
// ---------------------------------------------------------------------------
__global__ __launch_bounds__(256) void k_lfu(const float* __restrict__ fmean,
                                             const float* __restrict__ Wl,
                                             float* __restrict__ Y) {
  int gid = blockIdx.x * 256 + threadIdx.x;  // 2*1024*128
  int o = gid & 127;
  int p = (gid >> 7) & 1023;
  int b = gid >> 17;
  int y = p >> 5, x = p & 31;
  float acc = 0.f;
  for (int dy = -2; dy <= 2; ++dy)
    for (int dx = -2; dx <= 2; ++dx) {
      int yy = refl(y + dy, 32), xx = refl(x + dx, 32);
      acc += Wl[o * 25 + (dy + 2) * 5 + (dx + 2)] * fmean[b * 1024 + yy * 32 + xx];
    }
  Y[(size_t)(b * 1024 + p) * 128 + o] = acc;
}

// ---------------------------------------------------------------------------
// pre_agg: 3x3 reflect conv, 256->128 channels at 32x32
// ---------------------------------------------------------------------------
__global__ __launch_bounds__(256) void k_agg(const float* __restrict__ kcat,
                                             const float* __restrict__ Wa,
                                             float* __restrict__ Y) {
  int gid = blockIdx.x * 256 + threadIdx.x;  // 2*1024*128
  int o = gid & 127;
  int p = (gid >> 7) & 1023;
  int b = gid >> 17;
  int y = p >> 5, x = p & 31;
  float acc = 0.f;
  for (int ky = 0; ky < 3; ++ky)
    for (int kx = 0; kx < 3; ++kx) {
      int yy = refl(y + ky - 1, 32), xx = refl(x + kx - 1, 32);
      const float* ip = kcat + (size_t)(b * 1024 + yy * 32 + xx) * 256;
      const float* wp = Wa + (size_t)o * 2304 + ky * 3 + kx;
      for (int c = 0; c < 256; ++c) acc += ip[c] * wp[c * 9];
    }
  Y[(size_t)(b * 1024 + p) * 128 + o] = acc;
}

// ---------------------------------------------------------------------------
// Windowed cross-attention with WMMA. One block (4 waves) per (b, head, cell).
// Q[64x32] * K^T[32x32pad] -> masked softmax -> A[64x32] * V[32x96]
// ---------------------------------------------------------------------------
__global__ __launch_bounds__(128) void k_attn(const float* __restrict__ Q,
                                              const float* __restrict__ K,
                                              const float* __restrict__ V,
                                              float* __restrict__ O) {
  __shared__ int widx[32];          // key cell index per window slot (-1 = masked)
  __shared__ uint32_t kfrag[512];   // B frags for score GEMM [t][lane][e]
  __shared__ uint32_t vlds[1536];   // V rows bf16 [j][96]
  __shared__ float sc[2048];        // scores [64][32]
  __shared__ uint32_t alds[1024];   // attn bf16 [64][32]

  const int tid = threadIdx.x;
  const int cell = blockIdx.x & 1023;
  const int head = (blockIdx.x >> 10) & 3;
  const int b = blockIdx.x >> 12;
  const int cy = cell >> 5, cx = cell & 31;

  if (tid < 32) {
    int w = -1;
    if (tid < 25) {
      int ry = cy + tid / 5 - 2, rx = cx + tid % 5 - 2;
      if (ry >= 0 && ry < 32 && rx >= 0 && rx < 32) w = ry * 32 + rx;
    }
    widx[tid] = w;
  }
  __syncthreads();

  for (int i = tid; i < 512; i += 128) {   // K^T fragments (zero padded slots)
    int e = i & 7, lane = (i >> 3) & 31, t = i >> 8;
    int h = lane >> 4;
    int j = t * 16 + (lane & 15);
    int w = widx[j];
    uint32_t pk = 0;
    if (w >= 0) {
      const float* kp = K + (size_t)(b * 1024 + w) * 128 + head * 32 + 2 * e + 16 * h;
      pk = pack2bf(kp[0], kp[1]);
    }
    kfrag[i] = pk;
  }
  for (int i = tid; i < 1536; i += 128) {  // V rows, NCHW gather
    int d2 = i % 48, j = i / 48;
    int w = widx[j];
    uint32_t pk = 0;
    if (w >= 0) {
      int c = head * 96 + 2 * d2;
      pk = pack2bf(V[(size_t)(b * 384 + c) * 1024 + w],
                   V[(size_t)(b * 384 + c + 1) * 1024 + w]);
    }
    vlds[i] = pk;
  }

  const int wave = tid >> 5, lane = tid & 31;
  const int h = lane >> 4;
  const int qi = wave * 16 + (lane & 15);
  const int qy = qi >> 3, qx = qi & 7;
  const int p = (cy * 8 + qy) * 256 + (cx * 8 + qx);
  FragB qa;
  #pragma unroll
  for (int e = 0; e < 8; ++e) {
    const float* qp = Q + (size_t)(b * 65536 + p) * 128 + head * 32 + a_kbase(e, h);
    qa.u[e] = pack2bf(qp[0], qp[1]);
  }
  __syncthreads();

  v8f s0v, s1v;
  #pragma unroll
  for (int v = 0; v < 8; ++v) { s0v[v] = 0.f; s1v[v] = 0.f; }
  {
    FragB f0, f1;
    #pragma unroll
    for (int e = 0; e < 8; ++e) {
      f0.u[e] = kfrag[lane * 8 + e];
      f1.u[e] = kfrag[256 + lane * 8 + e];
    }
    s0v = __builtin_amdgcn_wmma_f32_16x16x32_bf16(false, qa.v, false, f0.v,
                                                  (short)0, s0v, false, false);
    s1v = __builtin_amdgcn_wmma_f32_16x16x32_bf16(false, qa.v, false, f1.v,
                                                  (short)0, s1v, false, false);
  }
  const float scale = 0.17677669529663687f;  // 1/sqrt(32)
  #pragma unroll
  for (int v = 0; v < 8; ++v) {
    int m = wave * 16 + v + 8 * h;
    int n0 = lane & 15, n1 = 16 + (lane & 15);
    float a0 = (widx[n0] < 0) ? -1e9f : s0v[v] * scale;
    float a1 = (widx[n1] < 0) ? -1e9f : s1v[v] * scale;
    sc[m * 32 + n0] = a0;
    sc[m * 32 + n1] = a1;
  }
  __syncthreads();

  if (tid < 64) {
    float mx = -1e30f;
    for (int j = 0; j < 32; ++j) mx = fmaxf(mx, sc[tid * 32 + j]);
    float sum = 0.f;
    for (int j = 0; j < 32; ++j) sum += __expf(sc[tid * 32 + j] - mx);
    float inv = 1.f / sum;
    for (int j = 0; j < 32; j += 2)
      alds[tid * 16 + (j >> 1)] = pack2bf(__expf(sc[tid * 32 + j] - mx) * inv,
                                          __expf(sc[tid * 32 + j + 1] - mx) * inv);
  }
  __syncthreads();

  FragB aa;
  #pragma unroll
  for (int e = 0; e < 8; ++e)
    aa.u[e] = alds[(wave * 16 + (lane & 15)) * 16 + (a_kbase(e, h) >> 1)];

  v8f oacc[6];
  #pragma unroll
  for (int t = 0; t < 6; ++t)
    #pragma unroll
    for (int v = 0; v < 8; ++v) oacc[t][v] = 0.f;
  #pragma unroll
  for (int t = 0; t < 6; ++t) {
    FragB fb;
    #pragma unroll
    for (int e = 0; e < 8; ++e) {
      int j = 2 * e + 16 * h;
      int n = t * 16 + (lane & 15);
      uint32_t r0 = vlds[j * 48 + (n >> 1)];
      uint32_t r1 = vlds[(j + 1) * 48 + (n >> 1)];
      uint32_t lo = (n & 1) ? (r0 >> 16) : (r0 & 0xffffu);
      uint32_t hi = (n & 1) ? (r1 & 0xffff0000u) : (r1 << 16);
      fb.u[e] = lo | hi;
    }
    oacc[t] = __builtin_amdgcn_wmma_f32_16x16x32_bf16(false, aa.v, false, fb.v,
                                                      (short)0, oacc[t], false, false);
  }
  #pragma unroll
  for (int t = 0; t < 6; ++t) {
    int d = t * 16 + (lane & 15);
    #pragma unroll
    for (int v = 0; v < 8; ++v) {
      int m = wave * 16 + v + 8 * h;
      int pp = (cy * 8 + (m >> 3)) * 256 + (cx * 8 + (m & 7));
      O[(size_t)(b * 384 + head * 96 + d) * 65536 + pp] = oacc[t][v];
    }
  }
}

// ---------------------------------------------------------------------------
extern "C" void kernel_launch(void* const* d_in, const int* in_sizes, int n_in,
                              void* d_out, int out_size, void* d_ws, size_t ws_size,
                              hipStream_t stream) {
  (void)in_sizes; (void)n_in; (void)out_size; (void)ws_size;
  const float* image     = (const float*)d_in[0];
  const float* features  = (const float*)d_in[1];
  const float* pre_img   = (const float*)d_in[2];
  const float* pre_key   = (const float*)d_in[3];
  const float* pre_query = (const float*)d_in[4];
  const float* lfu_w     = (const float*)d_in[5];
  const float* pre_agg   = (const float*)d_in[6];
  const float* rb_w1     = (const float*)d_in[7];
  const float* rb_w2     = (const float*)d_in[8];
  const float* rb_gamma  = (const float*)d_in[9];
  const float* rb_beta   = (const float*)d_in[10];
  float* out = (float*)d_out;

  float* ws = (float*)d_ws;
  const size_t FB = (size_t)2 * 65536 * 128;   // full-res [B,P,128]
  float* B0 = ws;
  float* B1 = ws + FB;
  float* B2 = ws + 2 * FB;
  float* B3 = ws + 3 * FB;
  float* B4 = ws + 4 * FB;
  float* kcat = ws + 5 * FB;                   // [B,1024,256]
  float* S0 = kcat + (size_t)2 * 1024 * 256;   // [B,1024,128] x3
  float* S1 = S0 + (size_t)2 * 1024 * 128;
  float* S2 = S1 + (size_t)2 * 1024 * 128;
  float* FM = S2 + (size_t)2 * 1024 * 128;     // [B,1024]
  float* STM = FM + 2048;                      // GN mean [B*8]
  float* STV = STM + 16;                       // GN var  [B*8]

  auto gemm = [&](const float* X, const float* W, const float* R, float* Y,
                  int P, int act, const float* g, const float* be,
                  int ostr, int ooff) {
    k_gemm<<<dim3((2 * P) / 128), dim3(256), 0, stream>>>(
        X, W, R, Y, STM, STV, g, be, P, act, ostr, ooff);
  };
  auto rblock = [&](const float* Xin, float* H, float* Yout, int i, int P,
                    int ostr, int ooff) {
    gemm(Xin, rb_w1 + (size_t)i * 16384, nullptr, H, P, 0, nullptr, nullptr, 128, 0);
    k_gnstats<<<dim3(16), dim3(256), 0, stream>>>(H, STM, STV, P);
    gemm(H, rb_w2 + (size_t)i * 16384, Xin, Yout, P, 1,
         rb_gamma + (size_t)i * 128, rb_beta + (size_t)i * 128, ostr, ooff);
  };

  // encoder + RoPE
  k_preimg<<<dim3(65536), dim3(256), 0, stream>>>(image, pre_img, B0);
  rblock(B0, B1, B2, 0, 65536, 128, 0);
  rblock(B2, B1, B0, 1, 65536, 128, 0);
  k_rope<<<dim3(32768), dim3(256), 0, stream>>>(B0);       // enc = B0

  // query path (pool is identity at full res)
  gemm(B0, pre_query, nullptr, B1, 65536, 0, nullptr, nullptr, 128, 0);
  rblock(B1, B2, B3, 4, 65536, 128, 0);
  rblock(B3, B2, B4, 5, 65536, 128, 0);                    // q = B4

  // key path (full res then 8x8 pool)
  gemm(B0, pre_key, nullptr, B1, 65536, 0, nullptr, nullptr, 128, 0);
  rblock(B1, B2, B3, 2, 65536, 128, 0);
  rblock(B3, B2, B1, 3, 65536, 128, 0);                    // kimg = B1
  k_pool<<<dim3(1024), dim3(256), 0, stream>>>(B1, kcat);  // kcat[:, :128]

  // key-features path (LFU)
  k_featmean<<<dim3(8), dim3(256), 0, stream>>>(features, FM);
  k_lfu<<<dim3(1024), dim3(256), 0, stream>>>(FM, lfu_w, S0);
  rblock(S0, S1, S2, 6, 1024, 128, 0);
  rblock(S2, S1, kcat, 7, 1024, 256, 128);                 // kcat[:, 128:]

  // aggregation
  k_agg<<<dim3(1024), dim3(256), 0, stream>>>(kcat, pre_agg, S0);
  rblock(S0, S1, S2, 8, 1024, 128, 0);
  rblock(S2, S1, S0, 9, 1024, 128, 0);                     // k = S0

  // windowed cross-attention -> NCHW output
  k_attn<<<dim3(8192), dim3(128), 0, stream>>>(B4, S0, features, out);
}
